// Sparse2DBackbone_51402168598645
// MI455X (gfx1250) — compile-verified, matrix operand order checked by reading, not run on hardware
//
#include <hip/hip_runtime.h>
#include <hip/hip_bf16.h>
#include <stdint.h>

// ---------------------------------------------------------------------------
// Sparse 2D backbone for MI455X (gfx1250, wave32).
//  * every conv is an implicit GEMM through v_wmma_f32_16x16x32_f16
//  * full-res layers evaluated only at the 60000 active sites
//  * M=64 output tile (4 WMMA m-subtiles), N=128/256 across 8 waves
//  * weights pre-converted once to f16, chunk-contiguous [chunk][co][kk]
//  * A/B tiles double-buffered in LDS; staging uses the CDNA5 async-to-LDS
//    path (global_load_async_to_lds_b128 + s_wait_asynccnt), overlapping
//    loads with WMMA; tap/chunk addressing advances incrementally (no divs)
// ---------------------------------------------------------------------------

typedef __attribute__((ext_vector_type(16))) _Float16 v16h;
typedef __attribute__((ext_vector_type(8)))  _Float16 h8;
typedef __attribute__((ext_vector_type(8)))  float    v8f;

#define EPSV 1e-3f

// ------------------------------- utility kernels ---------------------------

__global__ void k_zero128(uint4* __restrict__ p, long long n) {
  long long i  = (long long)blockIdx.x * blockDim.x + threadIdx.x;
  long long st = (long long)gridDim.x * blockDim.x;
  uint4 z; z.x = 0u; z.y = 0u; z.z = 0u; z.w = 0u;
  for (; i < n; i += st) p[i] = z;
}

__global__ void k_filli(int* __restrict__ p, long long n, int v) {
  long long i  = (long long)blockIdx.x * blockDim.x + threadIdx.x;
  long long st = (long long)gridDim.x * blockDim.x;
  for (; i < n; i += st) p[i] = v;
}

// deterministic duplicate resolution: highest point index owns the site
__global__ void k_owner(const int* __restrict__ coords, int* __restrict__ owner, int n) {
  int s = blockIdx.x * blockDim.x + threadIdx.x;
  if (s >= n) return;
  int b = coords[s * 3 + 0], y = coords[s * 3 + 1], x = coords[s * 3 + 2];
  long long site = ((long long)b * 400 + y) * 400 + x;
  atomicMax(&owner[site], s);
}

__global__ void k_scatter(const float* __restrict__ feats, const int* __restrict__ coords,
                          const int* __restrict__ owner, _Float16* __restrict__ dst,
                          unsigned char* __restrict__ m0) {
  int s = blockIdx.x;
  int c = threadIdx.x;           // 128 threads = 128 channels
  int b = coords[s * 3 + 0], y = coords[s * 3 + 1], x = coords[s * 3 + 2];
  long long site = ((long long)b * 400 + y) * 400 + x;
  if (owner[site] != s) return;  // only the winning duplicate writes
  dst[site * 128 + c] = (_Float16)feats[(long long)s * 128 + c];
  if (c == 0) m0[site] = 1;
}

// mask1 = maxpool(mask0, 3x3, stride 2, pad 1) > 0
__global__ void k_maskdown(const unsigned char* __restrict__ m0, unsigned char* __restrict__ m1) {
  int p = blockIdx.x * blockDim.x + threadIdx.x;
  if (p >= 2 * 200 * 200) return;
  int b  = p / (200 * 200);
  int r  = p - b * (200 * 200);
  int py = r / 200, px = r - (r / 200) * 200;
  unsigned char v = 0;
#pragma unroll
  for (int dy = -1; dy <= 1; ++dy)
#pragma unroll
    for (int dx = -1; dx <= 1; ++dx) {
      int iy = py * 2 + dy, ix = px * 2 + dx;
      if (iy >= 0 && iy < 400 && ix >= 0 && ix < 400)
        v |= m0[((long long)b * 400 + iy) * 400 + ix];
    }
  m1[p] = v ? 1 : 0;
}

// one-time weight prep: f32 (3,3,CTOT,C) -> f16 chunk-contiguous layout
//   wh[((tap*NCH + ch)*C + co)*32 + kk]  with cin = ch*32 + kk
// (flip folds the transposed-conv kernel flip at prep time)
__global__ void k_prepw(const float* __restrict__ w, _Float16* __restrict__ wh,
                        int CTOT, int C, int flip) {
  long long i = (long long)blockIdx.x * blockDim.x + threadIdx.x;
  long long tot = 9LL * CTOT * C;
  if (i >= tot) return;
  int NCH = CTOT / 32;
  int kk = (int)(i & 31);
  long long r = i >> 5;            // (tap*NCH + ch)*C + co
  int co = (int)(r % C);
  long long q = r / C;             // tap*NCH + ch
  int ch  = (int)(q % NCH);
  int tap = (int)(q / NCH);
  int tapw = flip ? (8 - tap) : tap;
  int cin  = ch * 32 + kk;
  wh[i] = (_Float16)w[((long long)tapw * CTOT + cin) * C + co];
}

// ------------------------------- WMMA conv ---------------------------------
// Fragment load per CDNA5 ISA 7.12.2 (16-bit A, 16x32):
//   lanes 0..15  : row = lane,    K in {0..7} and {16..23}
//   lanes 16..31 : row = lane-16, K in {8..15} and {24..31}
// B is staged transposed ([co][k]) so the same addressing applies.
__device__ __forceinline__ v16h ld_frag(const _Float16* base, int kb0) {
  h8 lo = *(const h8*)(base + kb0);
  h8 hi = *(const h8*)(base + kb0 + 16);
  v16h r;
#pragma unroll
  for (int i = 0; i < 8; ++i) { r[i] = lo[i]; r[i + 8] = hi[i]; }
  return r;
}

__device__ __forceinline__ void async_b128(unsigned lds, const void* g) {
  asm volatile("global_load_async_to_lds_b128 %0, %1, off" :: "v"(lds), "v"(g) : "memory");
}

// mode 0: sparse subm conv (stride 1, pad 1), M-list = coords
// mode 1: dense conv at output res OHxOW, stride `stride`, pad 1, mask applied
// mode 2: sparse transposed conv (lhs_dilation 2, pad lo 1), M-list = coords
template <int NSUB>
__global__ __launch_bounds__(256, 2)
void k_conv(const _Float16* __restrict__ in0, const _Float16* __restrict__ in1,
            int inH, int inW, int cin0, int cin1,
            const _Float16* __restrict__ wh, const float* __restrict__ bnp,
            const float* __restrict__ ebias, const unsigned char* __restrict__ maskp,
            const _Float16* __restrict__ zpad,
            _Float16* __restrict__ outH, float* __restrict__ outF,
            const int* __restrict__ coords, int nPts,
            int OH, int OW, int mode, int stride) {
  constexpr int C = NSUB * 128;           // output channels
  constexpr int INFLIGHT = 1 + 2 * NSUB;  // async instrs per wave per chunk
  __shared__ _Float16 lA[2][64 * 32];     // A tile, M=64 (double buffered)
  __shared__ _Float16 lB[2][C * 32];      // B tile transposed [co][k]
  __shared__ int sC[64 * 3];              // (b, oy, ox) per tile row

  const int t    = threadIdx.x;
  const int wv   = t >> 5;
  const int lane = t & 31;
  const int half = lane >> 4;
  const int lrow = lane & 15;
  const int kb0  = half << 3;
  const int CTOT = cin0 + cin1;
  const int TOT  = 9 * (CTOT / 32);       // total K chunks

  if (t < 64) {
    int b, oy, ox;
    if (mode == 1) {
      int p = blockIdx.x * 64 + t;
      b = p / (OH * OW);
      int r = p - b * (OH * OW);
      oy = r / OW; ox = r - (r / OW) * OW;
    } else {
      int s = blockIdx.x * 64 + t;
      if (s >= nPts) s = nPts - 1;        // pad tile with last coord
      b = coords[s * 3 + 0]; oy = coords[s * 3 + 1]; ox = coords[s * 3 + 2];
    }
    sC[t * 3 + 0] = b; sC[t * 3 + 1] = oy; sC[t * 3 + 2] = ox;
  }
  __syncthreads();

  // this thread's A-staging slot: row am, 8 channels from akq (16B async)
  const int am  = t >> 2;
  const int akq = (t & 3) << 3;
  const int ab  = sC[am * 3 + 0];
  const int aoy = sC[am * 3 + 1];
  const int aox = sC[am * 3 + 2];
  const unsigned ldsA[2] = {(unsigned)(size_t)&lA[0][am * 32 + akq],
                            (unsigned)(size_t)&lA[1][am * 32 + akq]};
  const unsigned ldsB[2] = {(unsigned)(size_t)&lB[0][0],
                            (unsigned)(size_t)&lB[1][0]};

  // ---- incremental staging state (no divisions in the hot loop) ----
  int kyN = 0, kxN = 0;                   // tap being staged next
  int cN  = 0;                            // channel base of next chunk
  const _Float16* wbN = wh;               // next B chunk (contiguous)
  const _Float16* a0 = zpad;              // per-tap A base ptrs (akq folded)
  const _Float16* a1 = zpad;
  bool vN = false;

  auto tapSetup = [&]() {
    int iy, ix; bool valid;
    if (mode == 2) {                      // transposed conv, pad lo = 1
      int dy = aoy + kyN - 1, dx = aox + kxN - 1;
      valid = (dy >= 0) && (dx >= 0) && ((dy & 1) == 0) && ((dx & 1) == 0);
      iy = dy >> 1; ix = dx >> 1;
      valid = valid && (iy < inH) && (ix < inW);
    } else {
      iy = aoy * stride + kyN - 1;
      ix = aox * stride + kxN - 1;
      valid = (iy >= 0) && (iy < inH) && (ix >= 0) && (ix < inW);
    }
    vN = valid;
    if (valid) {
      long long pix = ((long long)ab * inH + iy) * inW + ix;
      a0 = in0 + pix * cin0 + akq;
      a1 = in1 ? (in1 + pix * cin1 + (akq - cin0)) : zpad;   // ptr for c>=cin0
    }
  };

  auto stage = [&](int sel) {
    // A: one 16B async copy per thread; invalid taps read the zero pad
    const _Float16* gp = zpad;
    if (vN) gp = (cN < cin0) ? (a0 + cN) : (a1 + cN);
    async_b128(ldsA[sel], gp);
    // B: contiguous prepped f16 chunk, 16B per thread per iter
#pragma unroll
    for (int e = 0; e < (C * 32) / (256 * 8); ++e) {
      int off = (t + e * 256) * 8;        // f16 elements (8 = 16 bytes)
      async_b128(ldsB[sel] + (unsigned)(off * 2), wbN + off);
    }
  };

  auto advance = [&]() {
    wbN += C * 32;
    cN += 32;
    if (cN == CTOT) {
      cN = 0;
      if (++kxN == 3) { kxN = 0; ++kyN; }
      tapSetup();
    }
  };

  v8f acc[4][NSUB];
#pragma unroll
  for (int i = 0; i < 4; ++i)
#pragma unroll
    for (int j = 0; j < NSUB; ++j) {
      v8f z = {0.f, 0.f, 0.f, 0.f, 0.f, 0.f, 0.f, 0.f};
      acc[i][j] = z;
    }

  tapSetup();
  stage(0);
  advance();
  for (int chk = 0; chk < TOT; ++chk) {
    const int sel = chk & 1;
    if (chk + 1 < TOT) {
      stage(sel ^ 1);
      advance();
      // in-order async completion: <=INFLIGHT outstanding => chunk chk landed
      asm volatile("s_wait_asynccnt %0" :: "n"(INFLIGHT) : "memory");
    } else {
      asm volatile("s_wait_asynccnt 0" ::: "memory");
    }
    __syncthreads();

    v16h bf[NSUB];
#pragma unroll
    for (int ns = 0; ns < NSUB; ++ns)
      bf[ns] = ld_frag(&lB[sel][(ns * 128 + wv * 16 + lrow) * 32], kb0);
#pragma unroll
    for (int ms = 0; ms < 4; ++ms) {
      v16h af = ld_frag(&lA[sel][(ms * 16 + lrow) * 32], kb0);
#pragma unroll
      for (int ns = 0; ns < NSUB; ++ns)
        acc[ms][ns] = __builtin_amdgcn_wmma_f32_16x16x32_f16(
            false, af, false, bf[ns], (short)0, acc[ms][ns], false, false);
    }
    __syncthreads();   // compute done before this buffer is re-staged
  }

  // ---- epilogue: fused BN + ReLU (+ mask), f16 dense and/or f32 rows ------
#pragma unroll
  for (int ns = 0; ns < NSUB; ++ns) {
    const int co  = ns * 128 + wv * 16 + lrow;
    const float g  = bnp[0 * C + co];
    const float be = bnp[1 * C + co];
    const float mu = bnp[2 * C + co];
    const float va = bnp[3 * C + co];
    const float sc = g * rsqrtf(va + EPSV);
    const float eb = ebias ? ebias[co] : 0.f;
    const float sh = be + (eb - mu) * sc;   // folds conv bias + BN shift
#pragma unroll
    for (int ms = 0; ms < 4; ++ms) {
#pragma unroll
      for (int j = 0; j < 8; ++j) {
        const int m  = ms * 16 + (half << 3) + j;  // C/D layout 7.12.2
        const int b  = sC[m * 3 + 0];
        const int oy = sC[m * 3 + 1];
        const int ox = sC[m * 3 + 2];
        float v = acc[ms][ns][j] * sc + sh;
        v = v > 0.f ? v : 0.f;
        const long long pix = ((long long)b * OH + oy) * OW + ox;
        if (mode == 1) v *= (float)maskp[pix];     // mask1 at half res
        if (outH) outH[pix * C + co] = (_Float16)v;
        if (outF) {
          int s = blockIdx.x * 64 + m;
          if (s < nPts) outF[(long long)s * C + co] = v;
        }
      }
    }
  }
}

// ------------------------------- launcher ----------------------------------

extern "C" void kernel_launch(void* const* d_in, const int* in_sizes, int n_in,
                              void* d_out, int out_size, void* d_ws, size_t ws_size,
                              hipStream_t stream) {
  const float* feats   = (const float*)d_in[0];
  const int*   coords  = (const int*)d_in[1];
  const float* c1b0_w  = (const float*)d_in[2];
  const float* c1b0_bn = (const float*)d_in[3];
  const float* c1b1_w  = (const float*)d_in[4];
  const float* c1b1_bn = (const float*)d_in[5];
  const float* c2b0_w  = (const float*)d_in[6];
  const float* c2b0_bn = (const float*)d_in[7];
  const float* c2b1_w  = (const float*)d_in[8];
  const float* c2b1_bn = (const float*)d_in[9];
  const float* up_w    = (const float*)d_in[10];
  const float* up_bias = (const float*)d_in[11];
  const float* up_bn   = (const float*)d_in[12];
  const float* p0_w    = (const float*)d_in[13];
  const float* p0_bn   = (const float*)d_in[14];
  const float* p1_w    = (const float*)d_in[15];
  const float* p1_bn   = (const float*)d_in[16];
  float* out = (float*)d_out;

  const int N_PTS = in_sizes[1] / 3;  // 60000

  // workspace layout
  const long long FE = 2LL * 400 * 400 * 128;   // full-res f16 grid elements
  const long long HE = 2LL * 200 * 200 * 256;   // half-res
  char* ws = (char*)d_ws;
  _Float16* buf0 = (_Float16*)ws;               // scatter in -> later p0 out
  _Float16* buf1 = buf0 + FE;                   // c1b0 out   -> later up out (y)
  _Float16* buf2 = buf1 + FE;                   // c1b1 out (lat)
  _Float16* h0   = buf2 + FE;                   // c2b0 out (half res, 256ch)
  _Float16* h1   = h0 + HE;                     // c2b1 out
  unsigned char* mask0 = (unsigned char*)(h1 + HE);
  unsigned char* mask1 = mask0 + 2 * 400 * 400;
  int* owner = (int*)(mask1 + 2 * 200 * 200);
  _Float16* zpad = (_Float16*)(owner + 2 * 400 * 400);   // 256B zero pad page
  _Float16* wh_c1b0 = zpad + 128;
  _Float16* wh_c1b1 = wh_c1b0 + 9 * 128 * 128;
  _Float16* wh_c2b0 = wh_c1b1 + 9 * 128 * 128;
  _Float16* wh_c2b1 = wh_c2b0 + 9 * 128 * 256;
  _Float16* wh_up   = wh_c2b1 + 9 * 256 * 256;
  _Float16* wh_p0   = wh_up   + 9 * 256 * 128;
  _Float16* wh_p1   = wh_p0   + 9 * 256 * 128;

  dim3 B256(256);
  // re-zero sparse-scatter targets, masks and pad page (deterministic replay)
  k_zero128<<<2048, B256, 0, stream>>>((uint4*)buf0, (3 * FE * 2) / 16);
  k_zero128<<<64,   B256, 0, stream>>>((uint4*)mask0, (2 * 400 * 400 + 2 * 200 * 200) / 16);
  k_zero128<<<1,    B256, 0, stream>>>((uint4*)zpad, 16);
  k_filli<<<1250, B256, 0, stream>>>(owner, 2LL * 400 * 400, -1);

  // one-time weight conversion to chunk-contiguous f16
  k_prepw<<<(9 * 128 * 128 + 255) / 256, B256, 0, stream>>>(c1b0_w, wh_c1b0, 128, 128, 0);
  k_prepw<<<(9 * 128 * 128 + 255) / 256, B256, 0, stream>>>(c1b1_w, wh_c1b1, 128, 128, 0);
  k_prepw<<<(9 * 128 * 256 + 255) / 256, B256, 0, stream>>>(c2b0_w, wh_c2b0, 128, 256, 0);
  k_prepw<<<(9 * 256 * 256 + 255) / 256, B256, 0, stream>>>(c2b1_w, wh_c2b1, 256, 256, 0);
  k_prepw<<<(9 * 256 * 128 + 255) / 256, B256, 0, stream>>>(up_w,   wh_up,   256, 128, 1);
  k_prepw<<<(9 * 256 * 128 + 255) / 256, B256, 0, stream>>>(p0_w,   wh_p0,   256, 128, 0);
  k_prepw<<<(9 * 128 * 128 + 255) / 256, B256, 0, stream>>>(p1_w,   wh_p1,   128, 128, 0);

  k_owner<<<(N_PTS + 255) / 256, B256, 0, stream>>>(coords, owner, N_PTS);
  k_scatter<<<N_PTS, dim3(128), 0, stream>>>(feats, coords, owner, buf0, mask0);
  k_maskdown<<<(2 * 200 * 200 + 255) / 256, B256, 0, stream>>>(mask0, mask1);

  const int MT_FULL = (N_PTS + 63) / 64;       // 938
  const int MT_HALF = (2 * 200 * 200) / 64;    // 1250

  // c1b0: subm 128->128 (sparse)
  k_conv<1><<<MT_FULL, B256, 0, stream>>>(buf0, nullptr, 400, 400, 128, 0,
      wh_c1b0, c1b0_bn, nullptr, nullptr, zpad, buf1, nullptr, coords, N_PTS, 400, 400, 0, 1);
  // c1b1: subm 128->128 (sparse) -> lat
  k_conv<1><<<MT_FULL, B256, 0, stream>>>(buf1, nullptr, 400, 400, 128, 0,
      wh_c1b1, c1b1_bn, nullptr, nullptr, zpad, buf2, nullptr, coords, N_PTS, 400, 400, 0, 1);
  // c2b0: stride-2 conv 128->256 (dense half res, mask1)
  k_conv<2><<<MT_HALF, B256, 0, stream>>>(buf2, nullptr, 400, 400, 128, 0,
      wh_c2b0, c2b0_bn, nullptr, mask1, zpad, h0, nullptr, nullptr, 0, 200, 200, 1, 2);
  // c2b1: subm 256->256 (dense half res, mask1)
  k_conv<2><<<MT_HALF, B256, 0, stream>>>(h0, nullptr, 200, 200, 256, 0,
      wh_c2b1, c2b1_bn, nullptr, mask1, zpad, h1, nullptr, nullptr, 0, 200, 200, 1, 1);
  // up: transposed conv 256->128 (sparse at coords, flip folded in prep, +bias)
  k_conv<1><<<MT_FULL, B256, 0, stream>>>(h1, nullptr, 200, 200, 256, 0,
      wh_up, up_bn, up_bias, nullptr, zpad, buf1, nullptr, coords, N_PTS, 400, 400, 2, 1);
  // p0: subm concat(y, lat) 256->128 (sparse)
  k_conv<1><<<MT_FULL, B256, 0, stream>>>(buf1, buf2, 400, 400, 128, 128,
      wh_p0, p0_bn, nullptr, nullptr, zpad, buf0, nullptr, coords, N_PTS, 400, 400, 0, 1);
  // p1: subm 128->128 (sparse), f32 rows straight into d_out (gather fused)
  k_conv<1><<<MT_FULL, B256, 0, stream>>>(buf0, nullptr, 400, 400, 128, 0,
      wh_p1, p1_bn, nullptr, nullptr, zpad, nullptr, out, coords, N_PTS, 400, 400, 0, 1);
  (void)n_in; (void)out_size; (void)ws_size;
}
